// TransformerEncoder_80152679678676
// MI455X (gfx1250) — matrix operand order repaired and checked by reading, not compiled, not run on hardware
//
#include <hip/hip_runtime.h>

#define TOKENS (16 * 577)
#define SEQ 577
#define C_DIM 768
#define HID 3072
#define NHEAD 12
#define DH 64

typedef __attribute__((ext_vector_type(16))) __bf16 v16bf;
typedef __attribute__((ext_vector_type(8))) float v8f;
typedef __attribute__((ext_vector_type(4))) unsigned int v4u;
typedef __attribute__((ext_vector_type(8))) int v8i;
typedef __attribute__((ext_vector_type(4))) int v4i;

// Tensor Data Mover availability (probe-confirmed builtin; arity differs by toolchain)
#ifndef USE_TDM
#if defined(__has_builtin)
#if __has_builtin(__builtin_amdgcn_tensor_load_to_lds) && __has_builtin(__builtin_amdgcn_s_wait_tensorcnt)
#define USE_TDM 1
#else
#define USE_TDM 0
#endif
#else
#define USE_TDM 0
#endif
#endif

union Frag16 {
    v16bf v;
    unsigned int u[8];
    unsigned short h[16];
};

__device__ __forceinline__ unsigned short f32_to_bf16(float f) {
    unsigned int u = __float_as_uint(f);
    u += 0x7FFFu + ((u >> 16) & 1u);   // round-to-nearest-even
    return (unsigned short)(u >> 16);
}

__device__ __forceinline__ float gelu_exact(float v) {
    return 0.5f * v * (1.0f + erff(v * 0.70710678118654752f));
}

// ---------------------------------------------------------------------------
// fp32 -> bf16 elementwise conversion (weights)
// ---------------------------------------------------------------------------
__global__ __launch_bounds__(256) void convert_bf16(const float* __restrict__ src,
                                                    unsigned short* __restrict__ dst,
                                                    int n) {
    for (int i = blockIdx.x * 256 + threadIdx.x; i < n; i += gridDim.x * 256)
        dst[i] = f32_to_bf16(src[i]);
}

// ---------------------------------------------------------------------------
// LayerNorm over C=768, one workgroup per token, bf16 output
// ---------------------------------------------------------------------------
__global__ __launch_bounds__(256) void layernorm_bf16(const float* __restrict__ x,
                                                      const float* __restrict__ g,
                                                      const float* __restrict__ bta,
                                                      unsigned short* __restrict__ out) {
    __shared__ float s1[256];
    __shared__ float s2[256];
    const int t = blockIdx.x;
    const int i = threadIdx.x;
    const float* xp = x + (size_t)t * C_DIM;
    float v0 = xp[i], v1 = xp[i + 256], v2 = xp[i + 512];
    s1[i] = v0 + v1 + v2;
    s2[i] = v0 * v0 + v1 * v1 + v2 * v2;
    __syncthreads();
    for (int off = 128; off > 0; off >>= 1) {
        if (i < off) { s1[i] += s1[i + off]; s2[i] += s2[i + off]; }
        __syncthreads();
    }
    const float mu = s1[0] * (1.0f / C_DIM);
    const float var = s2[0] * (1.0f / C_DIM) - mu * mu;
    const float rs = rsqrtf(var + 1e-5f);
    unsigned short* op = out + (size_t)t * C_DIM;
    op[i]       = f32_to_bf16((v0 - mu) * rs * g[i]       + bta[i]);
    op[i + 256] = f32_to_bf16((v1 - mu) * rs * g[i + 256] + bta[i + 256]);
    op[i + 512] = f32_to_bf16((v2 - mu) * rs * g[i + 512] + bta[i + 512]);
}

// ---------------------------------------------------------------------------
// Tiled bf16 WMMA GEMM, double-buffered pipeline:
//   stage(k+1) [TDM for A, transposed gather for B] overlaps compute(k);
//   one barrier per k-step (TDM completion published via s_wait_tensorcnt
//   before the barrier signal).
//   MODE 0: +bias -> bf16 | MODE 1: +bias,GELU -> bf16 | MODE 2: +bias,+resid -> f32
// ---------------------------------------------------------------------------
#define SA_STR 40    // u16 per LDS row: 32 data + 8 pad (80B pitch, conflict-free)
#define SB_STR 40

template <int MODE>
__global__ __launch_bounds__(256) void gemm_bf16(const unsigned short* __restrict__ A,
                                                 const unsigned short* __restrict__ B,
                                                 const float* __restrict__ bias,
                                                 const float* __restrict__ resid,
                                                 unsigned short* __restrict__ outBf,
                                                 float* __restrict__ outF,
                                                 int M, int N, int K) {
    __shared__ alignas(16) unsigned short sA[2][128 * SA_STR];    // [m][k] padded
    __shared__ alignas(16) unsigned short sBt[2][128 * SB_STR];   // [n][k] padded

    const int tid = threadIdx.x;
    const int lane = tid & 31;
    const int wave = tid >> 5;
    const int laneHi = lane >> 4;
    const int ln = lane & 15;
    const int wm = wave >> 1;          // 0..3 : 32-row strip
    const int wn = wave & 1;           // 0..1 : 64-col strip
    const int bm = blockIdx.x * 128;
    const int bn = blockIdx.y * 128;

    v8f acc[2][4] = {};
    const int kTiles = K >> 5;

    auto stage_tile = [&](int kt, int buf) {
        const int k0 = kt << 5;
#if USE_TDM
        if (wave == 0) {
            // 2D TDM descriptor: tile 32(k) x 128(m), elem 2B, row stride K,
            // LDS pad 16B every 64B -> 80B pitch (== SA_STR).
            const unsigned long long gaddr =
                (unsigned long long)(const void*)(A + (size_t)bm * K + k0);
            const unsigned ldsA = (unsigned)(unsigned long long)(const void*)&sA[buf][0];
            const unsigned rowsRem = (unsigned)(M - bm);
            v4u g0 = {1u,                                   // count=1
                      ldsA,                                 // lds_addr
                      (unsigned)gaddr,                      // global_addr[31:0]
                      (unsigned)((gaddr >> 32) & 0x01FFFFFFull) | 0x80000000u}; // +type=2
            v8i g1 = {(int)0x06D10000u,                     // data_size=2B, pad 16B/64B
                      (int)(((unsigned)K & 0xFFFFu) << 16), // tensor_dim0 lo
                      (int)(((unsigned)K >> 16) | ((rowsRem & 0xFFFFu) << 16)),
                      (int)(32u << 16),                     // tile_dim0=32
                      (int)128u,                            // tile_dim1=128
                      (int)(unsigned)K,                     // dim0 stride lo
                      0, 0};
            v4i gz = {0, 0, 0, 0};
#if __clang_major__ >= 23
            v8i gz8 = {0, 0, 0, 0, 0, 0, 0, 0};
            __builtin_amdgcn_tensor_load_to_lds(g0, g1, gz, gz, gz8, 0);
#else
            __builtin_amdgcn_tensor_load_to_lds(g0, g1, gz, gz, 0);
#endif
        }
#else
        // manual A staging fallback: 512 uint4, 2 per thread, padded pitch
#pragma unroll
        for (int i = 0; i < 2; ++i) {
            int idx = tid * 2 + i;
            int row = idx >> 2, c4 = idx & 3;
            int gr = bm + row; if (gr > M - 1) gr = M - 1;
            *reinterpret_cast<uint4*>(&sA[buf][row * SA_STR + c4 * 8]) =
                *reinterpret_cast<const uint4*>(A + (size_t)gr * K + k0 + c4 * 8);
        }
#endif
        // B transposed gather: 512 tasks (col, k-group of 8); per task 8 u16
        // column loads (coalesced across lanes) packed into one ds_store_b128.
#pragma unroll
        for (int i = 0; i < 2; ++i) {
            const int t2 = tid * 2 + i;
            const int col = t2 & 127, kg = t2 >> 7;
            const unsigned short* gB = B + (size_t)(k0 + kg * 8) * N + bn + col;
            unsigned int w[4];
#pragma unroll
            for (int j = 0; j < 4; ++j) {
                const unsigned short lo = gB[(size_t)(2 * j) * N];
                const unsigned short hi = gB[(size_t)(2 * j + 1) * N];
                w[j] = (unsigned)lo | ((unsigned)hi << 16);
            }
            uint4 val; val.x = w[0]; val.y = w[1]; val.z = w[2]; val.w = w[3];
            *reinterpret_cast<uint4*>(&sBt[buf][col * SB_STR + kg * 8]) = val;
        }
        if (k0 + 64 < K)   // speculative prefetch two k-tiles ahead (B stream)
            __builtin_prefetch(B + (size_t)(k0 + 64 + (tid >> 7)) * N + bn + (tid & 127), 0, 1);
    };

    // prologue: fill buffer 0
    stage_tile(0, 0);
#if USE_TDM
    if (wave == 0) __builtin_amdgcn_s_wait_tensorcnt(0);
#endif
    __syncthreads();

    for (int kt = 0; kt < kTiles; ++kt) {
        const int buf = kt & 1;
        const bool more = (kt + 1) < kTiles;
        if (more) stage_tile(kt + 1, buf ^ 1);   // overlaps compute below

        Frag16 af[2];
        const unsigned int* pA = reinterpret_cast<const unsigned int*>(&sA[buf][0]);
#pragma unroll
        for (int tm = 0; tm < 2; ++tm) {
            const int rbase = (wm * 32 + tm * 16 + ln) * (SA_STR / 2);
#pragma unroll
            for (int p = 0; p < 8; ++p) {       // contiguous K pairs: k = 16*laneHi + e
                const int j = p + 8 * laneHi;
                af[tm].u[p] = pA[rbase + j];
            }
        }
        const unsigned int* pB = reinterpret_cast<const unsigned int*>(&sBt[buf][0]);
#pragma unroll
        for (int tn = 0; tn < 4; ++tn) {
            Frag16 bfr;
            const int cbase = (wn * 64 + tn * 16 + ln) * (SB_STR / 2) + 8 * laneHi;
#pragma unroll
            for (int p = 0; p < 8; ++p) bfr.u[p] = pB[cbase + p];  // 2x ds_load_b128
#pragma unroll
            for (int tm = 0; tm < 2; ++tm)
                acc[tm][tn] = __builtin_amdgcn_wmma_f32_16x16x32_bf16(
                    false, af[tm].v, false, bfr.v, (short)0, acc[tm][tn], false, false);
        }

        if (more) {
#if USE_TDM
            if (wave == 0) __builtin_amdgcn_s_wait_tensorcnt(0);
#endif
            __syncthreads();   // publish buf^1; everyone done reading buf
        }
    }

    // Epilogue: C layout row = r + 8*laneHi, col = lane&15
#pragma unroll
    for (int tm = 0; tm < 2; ++tm) {
#pragma unroll
        for (int tn = 0; tn < 4; ++tn) {
            const int col = bn + wn * 64 + tn * 16 + ln;
            const float bv = bias[col];
            const int r0 = bm + wm * 32 + tm * 16 + laneHi * 8;
#pragma unroll
            for (int r = 0; r < 8; ++r) {
                const int row = r0 + r;
                if (row < M) {
                    float v = acc[tm][tn][r] + bv;
                    if (MODE == 1) v = gelu_exact(v);
                    if (MODE == 2) {
                        v += resid[(size_t)row * N + col];
                        outF[(size_t)row * N + col] = v;
                    } else {
                        outBf[(size_t)row * N + col] = f32_to_bf16(v);
                    }
                }
            }
        }
    }
}

// ---------------------------------------------------------------------------
// Flash attention core. Softmax is computed in A-fragment layout: each lane
// owns one query row (16 of its 32 scores), so row max/sum are in-register
// reductions plus one shfl_xor(16) exchange; m/l are one scalar per lane.
// Only the O-rescale factor and final 1/l need C-layout broadcasts.
// ---------------------------------------------------------------------------
#define SK_STR 72     // u16: 64 data + 8 pad (144B pitch)
#define SV_STR 40     // u16: 32 data + 8 pad (80B pitch)
#define SP32_STR 36   // f32: 32 data + 4 pad (144B pitch)

template <bool MASK>
__device__ __forceinline__ void fa_block(const Frag16 (&qf)[2],
                                         float& mst, float& lst,
                                         v8f (&o)[4],
                                         const unsigned short* __restrict__ sK,
                                         const unsigned short* __restrict__ sVt,
                                         float* __restrict__ sp32,
                                         int kb, int lane) {
    const int laneHi = lane >> 4;
    const int ln = lane & 15;

    // S = Q K^T : 16 queries x 32 keys = two C tiles
    v8f s[2] = {};
    const unsigned int* pK = reinterpret_cast<const unsigned int*>(sK);
#pragma unroll
    for (int t = 0; t < 2; ++t)
#pragma unroll
        for (int kk = 0; kk < 2; ++kk) {
            Frag16 bK;
            const int kbase = (t * 16 + ln) * (SK_STR / 2) + kk * 16 + 8 * laneHi;
#pragma unroll
            for (int p = 0; p < 8; ++p) bK.u[p] = pK[kbase + p];
            s[t] = __builtin_amdgcn_wmma_f32_16x16x32_bf16(
                false, qf[kk].v, false, bK.v, (short)0, s[t], false, false);
        }
    // scale + spill C layout -> per-wave [row][col] f32 scratch
#pragma unroll
    for (int t = 0; t < 2; ++t)
#pragma unroll
        for (int r = 0; r < 8; ++r)
            sp32[(r + 8 * laneHi) * SP32_STR + t * 16 + ln] = s[t][r] * 0.125f;

    // reload in A layout: lane holds row ln; element e -> k = e + 8*(e>=8) + 8*laneHi
    float a[16];
    {
        const float* rowp = sp32 + ln * SP32_STR + 8 * laneHi;
        const float4 q0 = *reinterpret_cast<const float4*>(rowp);        // k 0-3
        const float4 q1 = *reinterpret_cast<const float4*>(rowp + 4);    // k 4-7
        const float4 q2 = *reinterpret_cast<const float4*>(rowp + 16);   // k 16-19
        const float4 q3 = *reinterpret_cast<const float4*>(rowp + 20);   // k 20-23
        a[0] = q0.x; a[1] = q0.y; a[2]  = q0.z; a[3]  = q0.w;
        a[4] = q1.x; a[5] = q1.y; a[6]  = q1.z; a[7]  = q1.w;
        a[8] = q2.x; a[9] = q2.y; a[10] = q2.z; a[11] = q2.w;
        a[12] = q3.x; a[13] = q3.y; a[14] = q3.z; a[15] = q3.w;
    }
    if (MASK) {
#pragma unroll
        for (int e = 0; e < 16; ++e) {
            const int k = e + ((e >= 8) ? 8 : 0) + 8 * laneHi;
            if (kb * 32 + k > SEQ - 1) a[e] = -1.0e30f;
        }
    }
    // row max: in-register + one cross-half exchange
    float mx = a[0];
#pragma unroll
    for (int e = 1; e < 16; ++e) mx = fmaxf(mx, a[e]);
    mx = fmaxf(mx, __shfl_xor(mx, 16, 32));
    const float mn = fmaxf(mst, mx);
    // exp + row sum
    float sm = 0.0f;
#pragma unroll
    for (int e = 0; e < 16; ++e) { a[e] = __expf(a[e] - mn); sm += a[e]; }
    sm += __shfl_xor(sm, 16, 32);
    const float f = __expf(mst - mn);
    lst = lst * f + sm;
    mst = mn;
    // P fragment: pack bf16 pairs in registers (already A layout)
    Frag16 pf;
#pragma unroll
    for (int p = 0; p < 8; ++p)
        pf.u[p] = (unsigned)f32_to_bf16(a[2 * p]) |
                  ((unsigned)f32_to_bf16(a[2 * p + 1]) << 16);
    // broadcast rescale factor to C-layout rows; rescale O
#pragma unroll
    for (int r = 0; r < 8; ++r) {
        const float fr = __shfl(f, r + 8 * laneHi, 32);
#pragma unroll
        for (int t = 0; t < 4; ++t) o[t][r] *= fr;
    }
    // O += P V : four 16-col tiles over Dh=64, b128 fragments
    const unsigned int* pV = reinterpret_cast<const unsigned int*>(sVt);
#pragma unroll
    for (int t = 0; t < 4; ++t) {
        Frag16 bV;
        const int vbase = (t * 16 + ln) * (SV_STR / 2);
#pragma unroll
        for (int p = 0; p < 8; ++p) bV.u[p] = pV[vbase + ((p < 4) ? (p + 4 * laneHi) : (p + 4 + 4 * laneHi))];
        o[t] = __builtin_amdgcn_wmma_f32_16x16x32_bf16(
            false, pf.v, false, bV.v, (short)0, o[t], false, false);
    }
}

// ---------------------------------------------------------------------------
// Flash attention driver: grid = B*H*3 (each WG owns 13 of the 37 query
// tiles); each wave holds 2 query tiles, K/V stream through LDS once per WG.
// K tile staged by TDM (OOB keys zero-filled, LDS pad -> 144B pitch).
// ---------------------------------------------------------------------------
#define TPW 2

__global__ __launch_bounds__(256) void attention_fa(const unsigned short* __restrict__ qkv,
                                                    unsigned short* __restrict__ y) {
    __shared__ alignas(16) unsigned short sK[32 * SK_STR];      // [key][dh]
    __shared__ alignas(16) unsigned short sVt[64 * SV_STR];     // [dh][key]
    __shared__ alignas(16) float sP32[8][16 * SP32_STR];        // per-wave scratch

    const int tid = threadIdx.x;
    const int lane = tid & 31;
    const int wave = tid >> 5;
    const int laneHi = lane >> 4;
    const int ln = lane & 15;

    const int bid = blockIdx.x;
    const int b = bid / (NHEAD * 3);
    const int rem = bid % (NHEAD * 3);
    const int h = rem / 3;
    const int third = rem % 3;
    const int tok0 = b * SEQ;
    const int qoff = h * DH;
    const int koff = C_DIM + h * DH;
    const int voff = 2 * C_DIM + h * DH;

    // this wave's query tiles: qb = third*13 + wave*TPW + j
    bool wk[TPW];
    Frag16 qf[TPW][2] = {};
    float mst[TPW], lst[TPW];
    v8f o[TPW][4] = {};
#pragma unroll
    for (int j = 0; j < TPW; ++j) {
        const int slot = wave * TPW + j;
        const int qb = third * 13 + slot;
        wk[j] = (slot < 13) && (qb * 16 < SEQ);
        mst[j] = -3.0e38f; lst[j] = 0.0f;
        if (wk[j]) {
            int qrow = qb * 16 + ln; if (qrow > SEQ - 1) qrow = SEQ - 1;
            const unsigned int* pq = reinterpret_cast<const unsigned int*>(
                qkv + (size_t)(tok0 + qrow) * (3 * C_DIM) + qoff);
#pragma unroll
            for (int kk = 0; kk < 2; ++kk)
#pragma unroll
                for (int p = 0; p < 8; ++p) {
                    const int jj = p + 8 * laneHi;
                    qf[j][kk].u[p] = pq[kk * 16 + jj];
                }
        }
    }

    for (int kb = 0; kb < 19; ++kb) {       // ceil(577/32) key blocks
        __syncthreads();                    // readers done with sK/sVt
#if USE_TDM
        if (wave == 0) {
            // K tile via TDM: tile 64(dh) x 32(keys), row stride 3*C, elem 2B,
            // pad 16B per 128B row -> 144B pitch (== SK_STR); key tail zero-filled.
            const unsigned long long gaddr = (unsigned long long)(const void*)(
                qkv + (size_t)(tok0 + kb * 32) * (3 * C_DIM) + koff);
            const unsigned ldsK = (unsigned)(unsigned long long)(const void*)&sK[0];
            const unsigned krem = (unsigned)(SEQ - kb * 32);
            v4u g0 = {1u, ldsK, (unsigned)gaddr,
                      (unsigned)((gaddr >> 32) & 0x01FFFFFFull) | 0x80000000u};
            v8i g1 = {(int)0x07110000u,        // 2B elems, pad 16B per 32 DW
                      (int)(64u << 16),        // tensor_dim0 = 64
                      (int)((krem & 0xFFFFu) << 16),   // tensor_dim1 = keys left
                      (int)(64u << 16),        // tile_dim0 = 64
                      (int)32u,                // tile_dim1 = 32
                      (int)(3 * C_DIM),        // dim0 stride = 2304
                      0, 0};
            v4i gz = {0, 0, 0, 0};
#if __clang_major__ >= 23
            v8i gz8 = {0, 0, 0, 0, 0, 0, 0, 0};
            __builtin_amdgcn_tensor_load_to_lds(g0, g1, gz, gz, gz8, 0);
#else
            __builtin_amdgcn_tensor_load_to_lds(g0, g1, gz, gz, 0);
#endif
        }
#else
        {   // manual K staging fallback: one uint4 per thread, row-major
            const int row = tid >> 3, c4 = tid & 7;
            int key = kb * 32 + row; if (key > SEQ - 1) key = SEQ - 1;
            *reinterpret_cast<uint4*>(sK + row * SK_STR + c4 * 8) =
                *reinterpret_cast<const uint4*>(
                    qkv + (size_t)(tok0 + key) * (3 * C_DIM) + koff + c4 * 8);
        }
#endif
        {   // V staging transposed: 256 tasks (dh, key-group of 8)
            const int dh = tid & 63, kg = tid >> 6;
            unsigned int w[4];
#pragma unroll
            for (int j = 0; j < 4; ++j) {
                int k0 = kb * 32 + kg * 8 + 2 * j;
                int ka = (k0     > SEQ - 1) ? SEQ - 1 : k0;
                int kc = (k0 + 1 > SEQ - 1) ? SEQ - 1 : k0 + 1;
                const unsigned short lo = qkv[(size_t)(tok0 + ka) * (3 * C_DIM) + voff + dh];
                const unsigned short hi = qkv[(size_t)(tok0 + kc) * (3 * C_DIM) + voff + dh];
                w[j] = (unsigned)lo | ((unsigned)hi << 16);
            }
            uint4 val; val.x = w[0]; val.y = w[1]; val.z = w[2]; val.w = w[3];
            *reinterpret_cast<uint4*>(sVt + dh * SV_STR + kg * 8) = val;
        }
#if USE_TDM
        if (wave == 0) __builtin_amdgcn_s_wait_tensorcnt(0);
#endif
        __syncthreads();

        if (kb < 18) {
#pragma unroll
            for (int j = 0; j < TPW; ++j)
                if (wk[j])   // wave-uniform -> EXEC stays full for WMMA
                    fa_block<false>(qf[j], mst[j], lst[j], o[j], sK, sVt, sP32[wave], kb, lane);
        } else {
#pragma unroll
            for (int j = 0; j < TPW; ++j)
                if (wk[j])
                    fa_block<true>(qf[j], mst[j], lst[j], o[j], sK, sVt, sP32[wave], kb, lane);
        }
    }

#pragma unroll
    for (int j = 0; j < TPW; ++j) {
        if (!wk[j]) continue;
        const int qb = third * 13 + wave * TPW + j;
        const float linv = 1.0f / lst[j];   // per lane: row ln
#pragma unroll
        for (int r = 0; r < 8; ++r) {
            const float lr = __shfl(linv, r + 8 * laneHi, 32);
            const int row = qb * 16 + r + 8 * laneHi;
            if (row < SEQ) {
#pragma unroll
                for (int t = 0; t < 4; ++t)
                    y[(size_t)(tok0 + row) * C_DIM + h * DH + t * 16 + ln] =
                        f32_to_bf16(o[j][t][r] * lr);
            }
        }
    }
}

// ---------------------------------------------------------------------------
// Host orchestration
// ---------------------------------------------------------------------------
extern "C" void kernel_launch(void* const* d_in, const int* in_sizes, int n_in,
                              void* d_out, int out_size, void* d_ws, size_t ws_size,
                              hipStream_t stream) {
    (void)in_sizes; (void)n_in; (void)out_size; (void)ws_size;

    const float* x      = (const float*)d_in[0];
    const float* ln1_g  = (const float*)d_in[1];
    const float* ln1_b  = (const float*)d_in[2];
    const float* qkv_w  = (const float*)d_in[3];
    const float* qkv_b  = (const float*)d_in[4];
    const float* proj_w = (const float*)d_in[5];
    const float* proj_b = (const float*)d_in[6];
    const float* ln2_g  = (const float*)d_in[7];
    const float* ln2_b  = (const float*)d_in[8];
    const float* fc1_w  = (const float*)d_in[9];
    const float* fc1_b  = (const float*)d_in[10];
    const float* fc2_w  = (const float*)d_in[11];
    const float* fc2_b  = (const float*)d_in[12];

    const int M = TOKENS;   // 9232
    char* ws = (char*)d_ws;
    size_t off = 0;
    auto take = [&](size_t elems) {
        unsigned short* p = (unsigned short*)(ws + off);
        off = (off + elems * 2 + 255) & ~(size_t)255;
        return p;
    };
    unsigned short* wqkv = take((size_t)C_DIM * 3 * C_DIM);
    unsigned short* wprj = take((size_t)C_DIM * C_DIM);
    unsigned short* wf1  = take((size_t)C_DIM * HID);
    unsigned short* wf2  = take((size_t)HID * C_DIM);
    unsigned short* xn   = take((size_t)M * C_DIM);        // reused as y
    unsigned short* qkv  = take((size_t)M * 3 * C_DIM);    // reused as xn2
    unsigned short* hbuf = take((size_t)M * HID);
    float* x1 = (float*)d_out;   // residual stream lives in d_out

    convert_bf16<<<2048, 256, 0, stream>>>(qkv_w,  wqkv, C_DIM * 3 * C_DIM);
    convert_bf16<<<2048, 256, 0, stream>>>(proj_w, wprj, C_DIM * C_DIM);
    convert_bf16<<<2048, 256, 0, stream>>>(fc1_w,  wf1,  C_DIM * HID);
    convert_bf16<<<2048, 256, 0, stream>>>(fc2_w,  wf2,  HID * C_DIM);

    layernorm_bf16<<<M, 256, 0, stream>>>(x, ln1_g, ln1_b, xn);

    dim3 gq((M + 127) / 128, (3 * C_DIM) / 128);
    gemm_bf16<0><<<gq, 256, 0, stream>>>(xn, wqkv, qkv_b, nullptr, qkv, nullptr,
                                         M, 3 * C_DIM, C_DIM);

    attention_fa<<<16 * NHEAD * 3, 256, 0, stream>>>(qkv, /*y=*/xn);

    dim3 gp((M + 127) / 128, C_DIM / 128);
    gemm_bf16<2><<<gp, 256, 0, stream>>>(xn, wprj, proj_b, x, nullptr, x1,
                                         M, C_DIM, C_DIM);

    unsigned short* xn2 = qkv;   // qkv dead after attention
    layernorm_bf16<<<M, 256, 0, stream>>>(x1, ln2_g, ln2_b, xn2);

    dim3 g1((M + 127) / 128, HID / 128);
    gemm_bf16<1><<<g1, 256, 0, stream>>>(xn2, wf1, fc1_b, nullptr, hbuf, nullptr,
                                         M, HID, C_DIM);

    dim3 g2((M + 127) / 128, C_DIM / 128);
    gemm_bf16<2><<<g2, 256, 0, stream>>>(hbuf, wf2, fc2_b, x1, nullptr, x1,
                                         M, C_DIM, HID);
}
